// WindowMamba_34522947125650
// MI455X (gfx1250) — matrix operand order repaired and checked by reading, not compiled, not run on hardware
//
#include <hip/hip_runtime.h>

// ---------------------------------------------------------------------------
// WindowMamba fused kernel for gfx1250 (MI455X).
// One workgroup (256 threads = 8 wave32) per 16x16 window; all intermediates
// in ~286KB dynamic LDS; GEMMs via v_wmma_f32_16x16x32_bf16; window gather
// and inverse-partition scatter via Tensor Data Mover (one 8-channel slab
// descriptor per wave) when the builtins are available.
// ---------------------------------------------------------------------------

typedef __attribute__((ext_vector_type(16))) __bf16          v16bf;
typedef __attribute__((ext_vector_type(8)))  float           v8f;
typedef __attribute__((ext_vector_type(4)))  float           f32x4;
typedef __attribute__((ext_vector_type(8)))  unsigned short  v8u16;
typedef __attribute__((ext_vector_type(16))) unsigned short  v16u16;
typedef __attribute__((ext_vector_type(4)))  unsigned int    u32x4;
typedef __attribute__((ext_vector_type(4)))  int             i32x4;
typedef __attribute__((ext_vector_type(8)))  int             i32x8;

#define NWIN   2048     // 8 * 16 * 16 windows
#define LTOK   256      // tokens per window (16x16)
#define DMODEL 64
#define DIN    128
#define DSTATE 16
#define DTRANK 4

// LDS layout (bytes). All offsets 16B aligned.
#define OFF_SA    0          // xw bf16 [256][72]                 = 36864
#define OFF_SU    36864      // u bf16 [256][136] (alias: fp32 TDM staging
                             //  [64][256]=65536B / sOut staging)  = 69632
#define OFF_SZ    106496     // z->y bf16 [256][136]              = 69632
#define OFF_DBC   176128     // x_dbl f32 [256][48]               = 49152
#define OFF_WG1   225280     // W_in  bf16 [256][72]              = 36864
#define OFF_WG2   262144     // W_x   bf16 [48][136]              = 13056
#define OFF_WG3   275200     // W_out bf16 [64][136]              = 17408
#define SMEM_BYTES 292608

#if __has_builtin(__builtin_amdgcn_tensor_load_to_lds)
#define HAVE_TDM 1
#endif
#if __has_builtin(__builtin_amdgcn_tensor_store_from_lds)
#define HAVE_TDM_ST 1
#endif
#if defined(__has_include)
#if __has_include(<hip/amd_detail/amd_gfx1250_TDM.h>)
#define TDM_6ARG 1            // therock-10.0 headers => clang-23 6-arg builtin
#endif
#endif

static __device__ __forceinline__ unsigned short f2bf(float f) {
    unsigned u = __float_as_uint(f);
    u += 0x7FFFu + ((u >> 16) & 1u);           // RNE
    return (unsigned short)(u >> 16);
}
static __device__ __forceinline__ float bf2f(unsigned short s) {
    return __uint_as_float(((unsigned)s) << 16);
}

#if HAVE_TDM || HAVE_TDM_ST
// Build D# groups for an 8-channel window slab: 3D tile, fp32 elements,
// tile_dim0=16 (w, contiguous), tile_dim1=16 (h, stride 256),
// tile_dim2=8 (c, stride 65536). LDS image: packed [c][16][16] f32.
struct TdmDesc { u32x4 g0; i32x8 g1; i32x4 g2, g3; };
static __device__ __forceinline__ TdmDesc
make_slab_desc(unsigned long long gaddr, unsigned lds_addr) {
    TdmDesc t;
    t.g0[0] = 1u;                                    // count=1, user mode
    t.g0[1] = lds_addr;                              // lds_addr (bytes)
    t.g0[2] = (unsigned)(gaddr & 0xFFFFFFFFull);     // global_addr[31:0]
    t.g0[3] = (unsigned)((gaddr >> 32) & 0x01FFFFFFull) | (2u << 30); // type=2
    t.g1[0] = (int)(2u << 16);                       // data_size = 4B
    t.g1[1] = (int)(256u << 16);                     // tensor_dim0 = 256 (W)
    t.g1[2] = (int)(256u << 16);                     // tensor_dim1 = 256 (H)
    t.g1[3] = (int)(16u << 16);                      // tile_dim0 = 16 (j)
    t.g1[4] = (int)(16u | (8u << 16));               // tile_dim1=16, tile_dim2=8
    t.g1[5] = 256;                                   // tensor_dim0_stride = W
    t.g1[6] = 0;                                     // (stride0 hi, stride1 lo16)
    t.g1[7] = 1;                                     // tensor_dim1_stride = 65536
    t.g2[0] = 8; t.g2[1] = 0; t.g2[2] = 0; t.g2[3] = 0;  // tensor_dim2 = 8
    t.g3[0] = 0; t.g3[1] = 0; t.g3[2] = 0; t.g3[3] = 0;
    return t;
}
#endif

// A fragment: 16x32 bf16, row-major source [row][k], per ISA 7.12.2:
// lane<16: K {k0..k0+7} in V0-3, {k0+16..k0+23} in V4-7 ; lane>=16: +8.
static __device__ __forceinline__ v16bf ldA(const unsigned short* base,
                                            int r0, int k0, int stride, int lane) {
    const int M = lane & 15, hi = lane >> 4;
    const unsigned short* p = base + (r0 + M) * stride + k0 + 8 * hi;
    v8u16 lo = *(const v8u16*)p;
    v8u16 hp = *(const v8u16*)(p + 16);
    v16u16 t = __builtin_shufflevector(lo, hp, 0,1,2,3,4,5,6,7,8,9,10,11,12,13,14,15);
    return __builtin_bit_cast(v16bf, t);
}

// B fragment: 32x16 bf16 where B[k][n] = W[n][k], W row-major [n][k]:
// lane holds N = lane&15, K = k0 + 8*(lane>>4) + {0..7} and +16 + {0..7}.
static __device__ __forceinline__ v16bf ldB(const unsigned short* base,
                                            int n0, int k0, int stride, int lane) {
    const int N = lane & 15, hi = lane >> 4;
    const unsigned short* p = base + (n0 + N) * stride + k0 + 8 * hi;
    v8u16 lo = *(const v8u16*)p;
    v8u16 hp = *(const v8u16*)(p + 16);
    v16u16 t = __builtin_shufflevector(lo, hp, 0,1,2,3,4,5,6,7,8,9,10,11,12,13,14,15);
    return __builtin_bit_cast(v16bf, t);
}

static __device__ __forceinline__ v8f wmma_bf16(v16bf a, v16bf b, v8f c) {
    return __builtin_amdgcn_wmma_f32_16x16x32_bf16(false, a, false, b,
                                                   (short)0, c, false, false);
}

__global__ void __launch_bounds__(256, 1)
window_mamba_kernel(const float* __restrict__ x,
                    const float* __restrict__ W_in,
                    const float* __restrict__ conv_w,
                    const float* __restrict__ conv_b,
                    const float* __restrict__ W_x,
                    const float* __restrict__ W_dt,
                    const float* __restrict__ b_dt,
                    const float* __restrict__ A_log,
                    const float* __restrict__ Dvec,
                    const float* __restrict__ W_out,
                    float* __restrict__ out) {
    extern __shared__ char smem[];
    unsigned short* sA   = (unsigned short*)(smem + OFF_SA);
    unsigned short* sU   = (unsigned short*)(smem + OFF_SU);
    unsigned short* sZ   = (unsigned short*)(smem + OFF_SZ);
    float*          sDBC = (float*)(smem + OFF_DBC);
    unsigned short* Wg1  = (unsigned short*)(smem + OFF_WG1);
    unsigned short* Wg2  = (unsigned short*)(smem + OFF_WG2);
    unsigned short* Wg3  = (unsigned short*)(smem + OFF_WG3);
    float*          sXF  = (float*)(smem + OFF_SU);   // staging [c][l] (alias)

    const int n    = blockIdx.x;        // window id
    const int bb   = n >> 8;            // batch
    const int nh   = (n >> 4) & 15;
    const int nw   = n & 15;
    const int tid  = threadIdx.x;
    const int lane = tid & 31;
    const int wv   = tid >> 5;          // wave id 0..7

    if (tid == 0) { __builtin_prefetch(W_out, 0, 1); }

    // ---------------- Phase A: stage window + weights into LDS --------------
#if HAVE_TDM
    // Each wave DMAs its own 8-channel slab into sXF[c][l]; 8 tensor ops in
    // flight, each tracked by its wave's TENSORcnt.
    {
        unsigned long long gaddr = (unsigned long long)(const void*)
            (x + (((size_t)bb * 64 + wv * 8) * 256 + nh * 16) * 256 + nw * 16);
        unsigned lds_addr = (unsigned)(size_t)(void*)(smem + OFF_SU) + wv * 8192u;
        TdmDesc t = make_slab_desc(gaddr, lds_addr);
#if TDM_6ARG
        i32x8 gx = {};
        __builtin_amdgcn_tensor_load_to_lds(t.g0, t.g1, t.g2, t.g3, gx, 0);
#else
        __builtin_amdgcn_tensor_load_to_lds(t.g0, t.g1, t.g2, t.g3, 0);
#endif
    }
#endif
    // W_in (256,64) -> Wg1 [256][72]
    for (int idx = tid; idx < 256 * 64; idx += 256)
        Wg1[(idx >> 6) * 72 + (idx & 63)] = f2bf(W_in[idx]);
    // W_x (36,128) -> Wg2 [48][136], rows 36..47 zero
    for (int idx = tid; idx < 48 * 128; idx += 256) {
        int r = idx >> 7, c = idx & 127;
        Wg2[r * 136 + c] = (r < 36) ? f2bf(W_x[r * 128 + c]) : (unsigned short)0;
    }
    // W_out (64,128) -> Wg3 [64][136]
    for (int idx = tid; idx < 64 * 128; idx += 256)
        Wg3[(idx >> 7) * 136 + (idx & 127)] = f2bf(W_out[idx]);

#if HAVE_TDM
    __builtin_amdgcn_s_wait_tensorcnt(0);
    __syncthreads();
    // LDS transpose + convert: sXF[c][l] (fp32) -> sA[l][c] (bf16)
    for (int idx = tid; idx < 64 * 256; idx += 256) {
        int c = idx >> 8, l = idx & 255;
        sA[l * 72 + c] = f2bf(sXF[c * 256 + l]);
    }
#else
    // Fallback: direct strided gather from global
    for (int idx = tid; idx < 64 * 256; idx += 256) {
        int c = idx >> 8, l = idx & 255;
        int i = l >> 4, j = l & 15;
        float g = x[(((size_t)bb * 64 + c) * 256 + nh * 16 + i) * 256 + nw * 16 + j];
        sA[l * 72 + c] = f2bf(g);
    }
#endif
    __syncthreads();

    // ---------------- Phase B: GEMM1  xz = xw * W_in^T  (256x64 * 64x256) --
    for (int mt2 = 0; mt2 < 2; ++mt2) {
        const int r0 = (wv * 2 + mt2) * 16;
        v16bf a0 = ldA(sA, r0, 0, 72, lane);
        v16bf a1 = ldA(sA, r0, 32, 72, lane);
        for (int nt = 0; nt < 16; ++nt) {
            v16bf b0 = ldB(Wg1, nt * 16, 0, 72, lane);
            v16bf b1 = ldB(Wg1, nt * 16, 32, 72, lane);
            v8f c = {};
            c = wmma_bf16(a0, b0, c);
            c = wmma_bf16(a1, b1, c);
            const int cM = (lane >> 4) * 8;
            const int col = nt * 16 + (lane & 15);
#pragma unroll
            for (int v = 0; v < 8; ++v) {
                int l = r0 + cM + v;
                unsigned short bv = f2bf(c[v]);
                if (col < DIN) sU[l * 136 + col] = bv;           // xc (pre-conv)
                else           sZ[l * 136 + (col - DIN)] = bv;   // z
            }
        }
    }
    __syncthreads();

    // ---------------- Phase C: causal depthwise conv (DC=4) + SiLU, in-place
    {
        const int d    = tid & 127;
        const int half = tid >> 7;
        const int l0   = half * 128;
        const float w0 = conv_w[d * 4 + 0], w1 = conv_w[d * 4 + 1];
        const float w2 = conv_w[d * 4 + 2], w3 = conv_w[d * 4 + 3];
        const float cb = conv_b[d];
        float xm3 = 0.f, xm2 = 0.f, xm1 = 0.f;
        if (half) {                       // stash boundary before overwrite
            xm3 = bf2f(sU[(l0 - 3) * 136 + d]);
            xm2 = bf2f(sU[(l0 - 2) * 136 + d]);
            xm1 = bf2f(sU[(l0 - 1) * 136 + d]);
        }
        __syncthreads();
        for (int l = l0; l < l0 + 128; ++l) {
            float xc = bf2f(sU[l * 136 + d]);
            float acc = cb + w0 * xm3 + w1 * xm2 + w2 * xm1 + w3 * xc;
            float u = acc / (1.f + __expf(-acc));               // SiLU
            sU[l * 136 + d] = f2bf(u);
            xm3 = xm2; xm2 = xm1; xm1 = xc;
        }
    }
    __syncthreads();

    // ---------------- Phase D: GEMM2  x_dbl = u * W_x^T (256x128 * 128x48) -
    for (int mt2 = 0; mt2 < 2; ++mt2) {
        const int r0 = (wv * 2 + mt2) * 16;
        v16bf a[4];
#pragma unroll
        for (int kk = 0; kk < 4; ++kk) a[kk] = ldA(sU, r0, kk * 32, 136, lane);
        for (int nt = 0; nt < 3; ++nt) {
            v8f c = {};
#pragma unroll
            for (int kk = 0; kk < 4; ++kk)
                c = wmma_bf16(a[kk], ldB(Wg2, nt * 16, kk * 32, 136, lane), c);
            const int cM = (lane >> 4) * 8;
            const int col = nt * 16 + (lane & 15);
#pragma unroll
            for (int v = 0; v < 8; ++v)
                sDBC[(r0 + cM + v) * 48 + col] = c[v];
        }
    }
    __syncthreads();

    // ---------------- Phase E: selective scan (sequential over L=256) -------
    // thread -> (channel d = tid>>1, states sh..sh+7). y gated in place over z.
    {
        const int d  = tid >> 1;
        const int sh = (tid & 1) * 8;
        float A8[8];
#pragma unroll
        for (int s = 0; s < 8; ++s) A8[s] = -__expf(A_log[d * 16 + sh + s]);
        const float wd0 = W_dt[d * 4 + 0], wd1 = W_dt[d * 4 + 1];
        const float wd2 = W_dt[d * 4 + 2], wd3 = W_dt[d * 4 + 3];
        const float bdt = b_dt[d];
        const float Dd  = Dvec[d];
        float h[8];
#pragma unroll
        for (int s = 0; s < 8; ++s) h[s] = 0.f;

        for (int l = 0; l < LTOK; ++l) {
            const float* rowp = sDBC + l * 48;
            float xdt = bdt + rowp[0] * wd0 + rowp[1] * wd1
                            + rowp[2] * wd2 + rowp[3] * wd3;
            float dt = (xdt > 20.f) ? xdt : log1pf(__expf(xdt));  // softplus
            float u  = bf2f(sU[l * 136 + d]);
            float du = dt * u;
            float y  = 0.f;
#pragma unroll
            for (int s = 0; s < 8; ++s) {
                float hv = h[s] * __expf(dt * A8[s]) + du * rowp[4 + sh + s];
                h[s] = hv;
                y += hv * rowp[20 + sh + s];
            }
            y += __shfl_xor(y, 1, 32);                  // sum the state pair
            float zv = bf2f(sZ[l * 136 + d]);
            float g  = zv / (1.f + __expf(-zv));        // SiLU(z)
            float yo = (y + u * Dd) * g;
            if ((tid & 1) == 0) sZ[l * 136 + d] = f2bf(yo);   // y over z
        }
    }
    __syncthreads();

    // ---------------- Phase F: GEMM3  out = y * W_out^T (256x128 * 128x64) -
#if HAVE_TDM_ST
    float* sOutC = (float*)(smem + OFF_SU);   // packed [c][l] f32 for TDM store
#else
    float* sOutF = (float*)(smem + OFF_SU);   // [l][c] staging, stride 68
#endif
    for (int mt2 = 0; mt2 < 2; ++mt2) {
        const int r0 = (wv * 2 + mt2) * 16;
        v16bf a[4];
#pragma unroll
        for (int kk = 0; kk < 4; ++kk) a[kk] = ldA(sZ, r0, kk * 32, 136, lane);
        for (int nt = 0; nt < 4; ++nt) {
            v8f c = {};
#pragma unroll
            for (int kk = 0; kk < 4; ++kk)
                c = wmma_bf16(a[kk], ldB(Wg3, nt * 16, kk * 32, 136, lane), c);
            const int cM = (lane >> 4) * 8;
            const int col = nt * 16 + (lane & 15);
#if HAVE_TDM_ST
            // lane's 8 accumulators are contiguous in l: two b128 LDS stores
            float* p = sOutC + col * 256 + r0 + cM;
            f32x4 lo; lo[0] = c[0]; lo[1] = c[1]; lo[2] = c[2]; lo[3] = c[3];
            f32x4 hi; hi[0] = c[4]; hi[1] = c[5]; hi[2] = c[6]; hi[3] = c[7];
            *(f32x4*)p = lo;
            *(f32x4*)(p + 4) = hi;
#else
#pragma unroll
            for (int v = 0; v < 8; ++v)
                sOutF[(r0 + cM + v) * 68 + col] = c[v];
#endif
        }
    }
    __syncthreads();

    // ---------------- Phase G: inverse window partition ---------------------
#if HAVE_TDM_ST
    // Each wave DMAs its 8-channel slab from sOutC straight into the strided
    // out[b,c,h,w] layout (same 3D descriptor as the load).
    {
        unsigned long long gaddr = (unsigned long long)(void*)
            (out + (((size_t)bb * 64 + wv * 8) * 256 + nh * 16) * 256 + nw * 16);
        unsigned lds_addr = (unsigned)(size_t)(void*)(smem + OFF_SU) + wv * 8192u;
        TdmDesc t = make_slab_desc(gaddr, lds_addr);
#if TDM_6ARG
        i32x8 gx = {};
        __builtin_amdgcn_tensor_store_from_lds(t.g0, t.g1, t.g2, t.g3, gx, 0);
#else
        __builtin_amdgcn_tensor_store_from_lds(t.g0, t.g1, t.g2, t.g3, 0);
#endif
        __builtin_amdgcn_s_wait_tensorcnt(0);
    }
#else
    for (int idx = tid; idx < 64 * 256; idx += 256) {
        int c = idx >> 8, l = idx & 255;
        int i = l >> 4, j = l & 15;
        out[(((size_t)bb * 64 + c) * 256 + nh * 16 + i) * 256 + nw * 16 + j] =
            sOutF[l * 68 + c];
    }
#endif
}

extern "C" void kernel_launch(void* const* d_in, const int* in_sizes, int n_in,
                              void* d_out, int out_size, void* d_ws, size_t ws_size,
                              hipStream_t stream) {
    (void)in_sizes; (void)n_in; (void)out_size; (void)d_ws; (void)ws_size;
    const float* x      = (const float*)d_in[0];
    const float* W_in   = (const float*)d_in[1];
    const float* conv_w = (const float*)d_in[2];
    const float* conv_b = (const float*)d_in[3];
    const float* W_x    = (const float*)d_in[4];
    const float* W_dt   = (const float*)d_in[5];
    const float* b_dt   = (const float*)d_in[6];
    const float* A_log  = (const float*)d_in[7];
    const float* Dvec   = (const float*)d_in[8];
    const float* W_out  = (const float*)d_in[9];
    float* out = (float*)d_out;

    (void)hipFuncSetAttribute((const void*)window_mamba_kernel,
                              hipFuncAttributeMaxDynamicSharedMemorySize,
                              SMEM_BYTES);
    window_mamba_kernel<<<NWIN, 256, SMEM_BYTES, stream>>>(
        x, W_in, conv_w, conv_b, W_x, W_dt, b_dt, A_log, Dvec, W_out, out);
}
